// GroupedQueryAttention_10746008174607
// MI455X (gfx1250) — matrix operand order repaired and checked by reading, not compiled, not run on hardware
//
#include <hip/hip_runtime.h>
#include <hip/hip_bf16.h>

// GQA attention for MI455X (gfx1250, wave32, WMMA, async LDS copies).
// B=2, T=2048, D=2048, 32 Q heads / 8 KV heads, d_head=64, RoPE base 10000.

#define T_SEQ   2048
#define DMODEL  2048
#define NHEADS  32
#define NKV     8
#define DHEAD   64
#define BATCH   2

typedef __attribute__((ext_vector_type(16))) _Float16 v16h;
typedef __attribute__((ext_vector_type(8)))  _Float16 v8h;
typedef __attribute__((ext_vector_type(4)))  _Float16 v4h;
typedef __attribute__((ext_vector_type(8)))  float    v8f;
typedef __attribute__((ext_vector_type(4)))  float    v4f;
typedef __attribute__((ext_vector_type(4)))  int      v4i;

#if __has_builtin(__builtin_amdgcn_global_load_async_to_lds_b128)
#define HAVE_ASYNC_LDS 1
#else
#define HAVE_ASYNC_LDS 0
#endif

// 16-byte global -> LDS copy. Async (ASYNCcnt-tracked, bypasses VGPRs) when
// the gfx1250 builtin is available, else a synchronous VGPR round trip.
// This clang types the builtin as (v4i __device__*, v4i __shared__*, Ii, Ii).
__device__ __forceinline__ void stage16(const _Float16* g, _Float16* l) {
#if HAVE_ASYNC_LDS
  __builtin_amdgcn_global_load_async_to_lds_b128(
      (__attribute__((address_space(1))) v4i*)(g),
      (__attribute__((address_space(3))) v4i*)(l), 0, 0);
#else
  *(v8h*)l = *(const v8h*)g;
#endif
}
__device__ __forceinline__ void async_wait_all() {
#if HAVE_ASYNC_LDS
  asm volatile("s_wait_asynccnt 0x0" ::: "memory");
#endif
}
__device__ __forceinline__ void async_wait_keep4() {
#if HAVE_ASYNC_LDS
  asm volatile("s_wait_asynccnt 0x4" ::: "memory");
#endif
}

__device__ __forceinline__ v8f wmma16(v16h a, v16h b, v8f c) {
  // (neg_a, A, neg_b, B, c_mod, C, reuse_a, reuse_b)
  return __builtin_amdgcn_wmma_f32_16x16x32_f16(false, a, false, b, (short)0, c,
                                                false, false);
}

// A-matrix fragment (16x32 f16): lane m = lane&15;
// lanes 0-15 hold K pairs {0..7, 16..23}, lanes 16-31 hold {8..15, 24..31}.
__device__ __forceinline__ v16h fragA_ld(const _Float16* p, int stride, int k0, int lane) {
  const int m  = lane & 15;
  const int hi = (lane >> 4) & 1;
  const _Float16* r = p + (size_t)m * stride + k0 + hi * 8;
  v8h lo = *(const v8h*)(r);
  v8h hh = *(const v8h*)(r + 16);
  return __builtin_shufflevector(lo, hh, 0,1,2,3,4,5,6,7,8,9,10,11,12,13,14,15);
}

// B-matrix fragment (32x16 f16), stored as B^T rows (n-major, contiguous in K):
// lane n = lane&15; lanes 0-15 hold K {0..7, 16..23}, lanes 16-31 hold {8..15, 24..31}.
__device__ __forceinline__ v16h fragB_ld(const _Float16* p, int stride, int k0, int lane) {
  const int n  = lane & 15;
  const int hi = (lane >> 4) & 1;
  const _Float16* r = p + (size_t)n * stride + k0 + hi * 8;
  v8h lo = *(const v8h*)(r);
  v8h hh = *(const v8h*)(r + 16);
  return __builtin_shufflevector(lo, hh, 0,1,2,3,4,5,6,7,8,9,10,11,12,13,14,15);
}

// ---------------------------------------------------------------------------
// fp32 -> f16 bulk conversion (one-shot; lets all GEMM staging be raw copies).
// ---------------------------------------------------------------------------
__global__ void __launch_bounds__(256)
gqa_f32_to_f16(const float* __restrict__ src, _Float16* __restrict__ dst, int n4) {
  const int i = blockIdx.x * 256 + threadIdx.x;
  if (i >= n4) return;
  v4f d = *(const v4f*)(src + (size_t)i * 4);
  v4h h; h[0] = (_Float16)d[0]; h[1] = (_Float16)d[1];
         h[2] = (_Float16)d[2]; h[3] = (_Float16)d[3];
  *(v4h*)(dst + (size_t)i * 4) = h;
}

// ---------------------------------------------------------------------------
// Tiled WMMA GEMM: C[M,N] = A[M,K] * W[N,K]^T, all-f16 operands, f32 accum.
// Workgroup tile 128x128, 8 waves (2x4), each wave 64x32 (4x2 wmma tiles).
// Double-buffered LDS staging via async global->LDS b128 copies.
// ---------------------------------------------------------------------------

// Copy a 128x32-half tile (row-major, dst row stride 40 halves) to LDS.
// 512 16B chunks, 2 per thread -> 4 async instructions per wave per tile pair.
__device__ __forceinline__ void stage_tile(const _Float16* src, size_t ld,
                                           _Float16 (*dst)[40], int tid) {
  #pragma unroll
  for (int i = 0; i < 2; ++i) {
    const int c = tid + i * 256;
    const int row = c >> 2, seg = (c & 3) * 8;
    stage16(src + (size_t)row * ld + seg, &dst[row][seg]);
  }
}

template<bool OUT_F32>
__global__ void __launch_bounds__(256)
gqa_gemm_xwT(const _Float16* __restrict__ A, const _Float16* __restrict__ W,
             void* __restrict__ Cp, int M, int N, int K) {
  __shared__ __align__(16) _Float16 As[2][128][40];
  __shared__ __align__(16) _Float16 Ws[2][128][40];

  const int tid = threadIdx.x, lane = tid & 31, wave = tid >> 5;
  const int bm = blockIdx.y * 128, bn = blockIdx.x * 128;
  const int wm = (wave >> 2) * 64, wn = (wave & 3) * 32;

  v8f acc[4][2] = {};

  const int ksteps = K / 32;
  // prologue: stage k-step 0 into buffer 0
  stage_tile(A + (size_t)bm * K, K, As[0], tid);
  stage_tile(W + (size_t)bn * K, K, Ws[0], tid);

  for (int s = 0; s < ksteps; ++s) {
    const int cur = s & 1;
    if (s + 1 < ksteps) {
      const int k1 = (s + 1) * 32;
      stage_tile(A + (size_t)bm * K + k1, K, As[cur ^ 1], tid);
      stage_tile(W + (size_t)bn * K + k1, K, Ws[cur ^ 1], tid);
      async_wait_keep4();          // retire current stage, leave next in flight
    } else {
      async_wait_all();
    }
    __syncthreads();

    v16h af[4], bf[2];
    #pragma unroll
    for (int i = 0; i < 4; ++i) af[i] = fragA_ld(&As[cur][wm + i * 16][0], 40, 0, lane);
    #pragma unroll
    for (int j = 0; j < 2; ++j) bf[j] = fragB_ld(&Ws[cur][wn + j * 16][0], 40, 0, lane);
    #pragma unroll
    for (int i = 0; i < 4; ++i)
      #pragma unroll
      for (int j = 0; j < 2; ++j)
        acc[i][j] = wmma16(af[i], bf[j], acc[i][j]);
    __syncthreads();               // everyone done reading before overwrite
  }

  // C/D layout: VGPR r -> row (r + 8*hi), col = lane&15.
  const int hi = lane >> 4, nn = lane & 15;
  #pragma unroll
  for (int i = 0; i < 4; ++i)
    #pragma unroll
    for (int j = 0; j < 2; ++j) {
      const size_t gn = (size_t)(bn + wn + j * 16 + nn);
      #pragma unroll
      for (int r = 0; r < 8; ++r) {
        const size_t gm = (size_t)(bm + wm + i * 16 + r + hi * 8);
        if constexpr (OUT_F32)
          ((float*)Cp)[gm * N + gn] = acc[i][j][r];
        else
          ((_Float16*)Cp)[gm * N + gn] = (_Float16)acc[i][j][r];
      }
    }
}

// ---------------------------------------------------------------------------
// RoPE (Llama style, rotate-half) applied in place to f16 Q or K.
// ---------------------------------------------------------------------------
__global__ void __launch_bounds__(256)
gqa_rope(_Float16* __restrict__ X, int nheads, int total) {
  const int idx = blockIdx.x * 256 + threadIdx.x;
  if (idx >= total) return;
  const int j    = idx & 31;
  const int rest = idx >> 5;
  const int hh   = rest % nheads;
  const int tok  = rest / nheads;
  const int t    = tok & (T_SEQ - 1);
  const float freq = __expf(-(float)j * (9.210340371976184f / 32.0f)); // 10000^(-j/32)
  float sn, cs;
  __sincosf((float)t * freq, &sn, &cs);
  _Float16* p = X + (size_t)tok * (size_t)(nheads * DHEAD) + hh * DHEAD + j;
  const float a = (float)p[0], b = (float)p[32];
  p[0]  = (_Float16)(a * cs - b * sn);
  p[32] = (_Float16)(b * cs + a * sn);
}

// ---------------------------------------------------------------------------
// Flash-style causal GQA attention. Grid (T/128, NHEADS, B), 256 threads.
// Each wave owns 16 query rows; workgroup stages 64-key K/V chunks in LDS.
// K chunk staged with async global->LDS; V transposed through VGPRs.
// ---------------------------------------------------------------------------
__global__ void __launch_bounds__(256)
gqa_flash(const _Float16* __restrict__ Q, const _Float16* __restrict__ Km,
          const _Float16* __restrict__ Vm, _Float16* __restrict__ Om) {
  __shared__ __align__(16) _Float16 Ks[64][72];      // [key][d]   (B^T for QK^T)
  __shared__ __align__(16) _Float16 Vt[64][72];      // [d][key]   (B^T for P*V)
  __shared__ __align__(16) _Float16 Ps[8][16][72];   // wave-private P (A source)

  const int tid = threadIdx.x, lane = tid & 31, w = tid >> 5;
  const int hi = lane >> 4, nn = lane & 15;
  const int qb = blockIdx.x * 128;
  const int h  = blockIdx.y, b = blockIdx.z;
  const int g  = h >> 2;                              // kv head = h / N_REP
  const int ldkv = NKV * DHEAD;                       // 512

  const _Float16* qbase = Q + (size_t)(b * T_SEQ + qb + w * 16) * DMODEL + h * DHEAD;
  const v16h qf0 = fragA_ld(qbase, DMODEL, 0,  lane);
  const v16h qf1 = fragA_ld(qbase, DMODEL, 32, lane);

  v8f oacc[4] = {};
  float mrow[8], lrow[8];
  #pragma unroll
  for (int r = 0; r < 8; ++r) { mrow[r] = -__builtin_inff(); lrow[r] = 0.0f; }

  const int nCh = qb / 64 + 2;                        // keys 0 .. qb+127
  for (int ch = 0; ch < nCh; ++ch) {
    const int kb = ch * 64;
    {
      // K chunk: 512 16B chunks, 2 per thread, async copy into LDS
      #pragma unroll
      for (int i = 0; i < 2; ++i) {
        const int c = tid + i * 256;
        const int row = c >> 3, seg = (c & 7) * 8;
        stage16(Km + (size_t)(b * T_SEQ + kb + row) * ldkv + g * DHEAD + seg,
                &Ks[row][seg]);
      }
      // V chunk transposed through VGPRs
      const int row = tid >> 2, seg = (tid & 3) * 16;
      const _Float16* vp = Vm + (size_t)(b * T_SEQ + kb + row) * ldkv + g * DHEAD + seg;
      v8h v0 = *(const v8h*)vp;
      v8h v1 = *(const v8h*)(vp + 8);
      #pragma unroll
      for (int i = 0; i < 8; ++i) { Vt[seg + i][row] = v0[i]; Vt[seg + 8 + i][row] = v1[i]; }
      // prefetch next chunk's K/V rows into cache while we compute
      if (ch + 1 < nCh) {
        const int prow = tid >> 2;
        __builtin_prefetch(Km + (size_t)(b * T_SEQ + kb + 64 + prow) * ldkv + g * DHEAD, 0, 1);
        __builtin_prefetch(Vm + (size_t)(b * T_SEQ + kb + 64 + prow) * ldkv + g * DHEAD, 0, 1);
      }
      async_wait_all();
    }
    __syncthreads();

    // S = Q * K^T : 16 q-rows x 64 keys (4 tiles x 2 k-steps)
    v8f s[4] = {};
    #pragma unroll
    for (int t = 0; t < 4; ++t) {
      v16h kf0 = fragB_ld(&Ks[t * 16][0], 72, 0,  lane);
      v16h kf1 = fragB_ld(&Ks[t * 16][0], 72, 32, lane);
      s[t] = wmma16(qf0, kf0, s[t]);
      s[t] = wmma16(qf1, kf1, s[t]);
    }

    // online softmax: rows r+8*hi live across the 16-lane half -> shfl_xor reduce
    #pragma unroll
    for (int r = 0; r < 8; ++r) {
      const int qrow = qb + w * 16 + r + hi * 8;
      float mx = -__builtin_inff();
      #pragma unroll
      for (int t = 0; t < 4; ++t) {
        const int kcol = kb + t * 16 + nn;
        float x = (kcol <= qrow) ? s[t][r] * 0.125f : -__builtin_inff();
        s[t][r] = x;
        mx = fmaxf(mx, x);
      }
      #pragma unroll
      for (int off = 8; off >= 1; off >>= 1) mx = fmaxf(mx, __shfl_xor(mx, off, 32));
      const float nm  = fmaxf(mrow[r], mx);
      const float esc = __expf(mrow[r] - nm);         // 0 on first chunk (m=-inf)
      float rs = 0.0f;
      #pragma unroll
      for (int t = 0; t < 4; ++t) {
        const float p = __expf(s[t][r] - nm);         // masked -> exp(-inf)=0
        rs += p;
        Ps[w][r + hi * 8][t * 16 + nn] = (_Float16)p; // C-layout -> LDS row-major
      }
      #pragma unroll
      for (int off = 8; off >= 1; off >>= 1) rs += __shfl_xor(rs, off, 32);
      lrow[r] = lrow[r] * esc + rs;
      mrow[r] = nm;
      #pragma unroll
      for (int d = 0; d < 4; ++d) oacc[d][r] *= esc;
    }

    // O += P * V  (re-read P in A layout; same-wave DS ops are in order)
    const v16h pf0 = fragA_ld(&Ps[w][0][0], 72, 0,  lane);
    const v16h pf1 = fragA_ld(&Ps[w][0][0], 72, 32, lane);
    #pragma unroll
    for (int d = 0; d < 4; ++d) {
      v16h vf0 = fragB_ld(&Vt[d * 16][0], 72, 0,  lane);
      v16h vf1 = fragB_ld(&Vt[d * 16][0], 72, 32, lane);
      oacc[d] = wmma16(pf0, vf0, oacc[d]);
      oacc[d] = wmma16(pf1, vf1, oacc[d]);
    }
    __syncthreads();
  }

  // normalize and write O as f16 in (B*T, NHEADS*DHEAD) layout
  #pragma unroll
  for (int r = 0; r < 8; ++r) {
    const float inv = 1.0f / lrow[r];
    const size_t tok = (size_t)(b * T_SEQ + qb + w * 16 + r + hi * 8);
    #pragma unroll
    for (int d = 0; d < 4; ++d)
      Om[tok * DMODEL + h * DHEAD + d * 16 + nn] = (_Float16)(oacc[d][r] * inv);
  }
}

// ---------------------------------------------------------------------------
extern "C" void kernel_launch(void* const* d_in, const int* in_sizes, int n_in,
                              void* d_out, int out_size, void* d_ws, size_t ws_size,
                              hipStream_t stream) {
  const float* x  = (const float*)d_in[0];
  const float* Wq = (const float*)d_in[1];
  const float* Wk = (const float*)d_in[2];
  const float* Wv = (const float*)d_in[3];
  const float* Wo = (const float*)d_in[4];

  const int M   = BATCH * T_SEQ;   // 4096
  const int K   = DMODEL;          // 2048
  const int Nq  = NHEADS * DHEAD;  // 2048
  const int Nkv = NKV * DHEAD;     // 512

  // workspace layout (f16 buffers), ~76 MB total
  char* ws = (char*)d_ws;
  size_t off = 0;
  _Float16* qb   = (_Float16*)(ws + off); off += (size_t)M * Nq  * 2;  // 16 MB
  _Float16* kb   = (_Float16*)(ws + off); off += (size_t)M * Nkv * 2;  //  4 MB
  _Float16* vb   = (_Float16*)(ws + off); off += (size_t)M * Nkv * 2;  //  4 MB
  _Float16* obuf = (_Float16*)(ws + off); off += (size_t)M * Nq  * 2;  // 16 MB
  _Float16* x16  = (_Float16*)(ws + off); off += (size_t)M * K   * 2;  // 16 MB
  _Float16* wq16 = (_Float16*)(ws + off); off += (size_t)Nq  * K * 2;  //  8 MB
  _Float16* wk16 = (_Float16*)(ws + off); off += (size_t)Nkv * K * 2;  //  2 MB
  _Float16* wv16 = (_Float16*)(ws + off); off += (size_t)Nkv * K * 2;  //  2 MB
  _Float16* wo16 = (_Float16*)(ws + off); off += (size_t)Nq  * K * 2;  //  8 MB

  dim3 blk(256);

  // one-shot fp32 -> f16 conversions (so GEMM staging is pure async copies)
  {
    const int nx = M * K / 4, nq = Nq * K / 4, nk = Nkv * K / 4;
    gqa_f32_to_f16<<<dim3((nx + 255) / 256), blk, 0, stream>>>(x,  x16,  nx);
    gqa_f32_to_f16<<<dim3((nq + 255) / 256), blk, 0, stream>>>(Wq, wq16, nq);
    gqa_f32_to_f16<<<dim3((nk + 255) / 256), blk, 0, stream>>>(Wk, wk16, nk);
    gqa_f32_to_f16<<<dim3((nk + 255) / 256), blk, 0, stream>>>(Wv, wv16, nk);
    gqa_f32_to_f16<<<dim3((nq + 255) / 256), blk, 0, stream>>>(Wo, wo16, nq);
  }

  // Q/K/V projections (f16 WMMA, f32 accumulate)
  gqa_gemm_xwT<false><<<dim3(Nq  / 128, M / 128), blk, 0, stream>>>(x16, wq16, qb, M, Nq,  K);
  gqa_gemm_xwT<false><<<dim3(Nkv / 128, M / 128), blk, 0, stream>>>(x16, wk16, kb, M, Nkv, K);
  gqa_gemm_xwT<false><<<dim3(Nkv / 128, M / 128), blk, 0, stream>>>(x16, wv16, vb, M, Nkv, K);

  // RoPE on Q and K
  {
    const int totq = M * NHEADS * 32;
    gqa_rope<<<dim3((totq + 255) / 256), blk, 0, stream>>>(qb, NHEADS, totq);
    const int totk = M * NKV * 32;
    gqa_rope<<<dim3((totk + 255) / 256), blk, 0, stream>>>(kb, NKV, totk);
  }

  // causal grouped attention
  gqa_flash<<<dim3(T_SEQ / 128, NHEADS, BATCH), blk, 0, stream>>>(qb, kb, vb, obuf);

  // output projection (f16 A, fp32 out)
  gqa_gemm_xwT<true><<<dim3(DMODEL / 128, M / 128), blk, 0, stream>>>(obuf, wo16, (float*)d_out, M, DMODEL, K);
}
